// MultiHeadAttention_6519760355431
// MI455X (gfx1250) — matrix operand order repaired and checked by reading, not compiled, not run on hardware
//
#include <hip/hip_runtime.h>

typedef __attribute__((ext_vector_type(16))) __bf16 v16bf;
typedef __attribute__((ext_vector_type(8)))  __bf16 v8bf;
typedef __attribute__((ext_vector_type(4)))  __bf16 v4bf;
typedef __attribute__((ext_vector_type(8)))  float  v8f;
typedef __attribute__((ext_vector_type(4)))  unsigned int u32x4;
typedef __attribute__((ext_vector_type(8)))  int      i32x8;
typedef __attribute__((ext_vector_type(4)))  int      i32x4;

#define B_  2
#define S_  2048
#define D_  1024
#define H_  16
#define DK_ 64
#define M_  (B_ * S_)          // 4096 rows total
static constexpr float LN_EPS = 1e-5f;

// Padded LDS layout for TDM-staged A bands: TDM inserts 4 DWORDs (8 halves)
// after every 256 DWORDs (512 halves).  Row stride 1040 halves -> rows hit
// different bank groups; all fragment loads stay 16B-aligned.
#define APAD_ROW 1040
__device__ __forceinline__ int apad_idx(int lin) { return lin + ((lin >> 9) << 3); }

union Frag {
    v16bf v;
    v8bf  h[2];
};

__device__ __forceinline__ v8bf load8(const __bf16* p) { return *(const v8bf*)p; }

__device__ __forceinline__ v8f zero8() {
    v8f z;
#pragma unroll
    for (int i = 0; i < 8; ++i) z[i] = 0.0f;
    return z;
}

__device__ __forceinline__ v8f wmma_bf16(const Frag& a, const Frag& b, v8f c) {
    // (neg_a, A, neg_b, B, c_mod, C, reuse_a, reuse_b)
    return __builtin_amdgcn_wmma_f32_16x16x32_bf16(false, a.v, false, b.v,
                                                   (short)0, c, false, false);
}

// ---------------------------------------------------------------------------
// Tensor Data Mover: DMA a 16 x 1024 bf16 band (row stride D_) into LDS with
// 4-DWORD padding every 256 DWORDs.  Descriptor per CDNA5 ISA.
// Issued by one wave; caller waits TENSORcnt==0 then barriers.
// ---------------------------------------------------------------------------
__device__ __forceinline__ void tdm_load_band16x1024(const __bf16* gsrc,
                                                     unsigned int lds_off) {
    const unsigned long long ga = (unsigned long long)(uintptr_t)gsrc;
    u32x4 g0;
    g0[0] = 1u;                                   // count=1, is_restore=0
    g0[1] = lds_off;                              // lds_addr (bytes)
    g0[2] = (unsigned int)ga;                     // global_addr[31:0]
    g0[3] = (unsigned int)((ga >> 32) & 0x1FFFFFFu) | (2u << 30); // [56:32]|type=2
    i32x8 g1;
    g1[0] = (1 << 16)        // data_size = 1 -> 2 bytes
          | (1 << 20)        // pad_enable
          | (7 << 22)        // pad_interval code 7 = 256 DWORDs
          | (3 << 25);       // pad_amount  code 3 = 4 DWORDs (16B, keeps align)
    g1[1] = (int)(1024u << 16);   // [63:48]  tensor_dim0 lo16 = 1024
    g1[2] = (int)(16u << 16);     // [79:64]  dim0 hi = 0 ; [95:80] tensor_dim1 = 16
    g1[3] = (int)(1024u << 16);   // [111:96] dim1 hi = 0 ; [127:112] tile_dim0 = 1024
    g1[4] = 16;                   // [143:128] tile_dim1 = 16 ; tile_dim2 = 0
    g1[5] = 1024;                 // tensor_dim0_stride lo32 (elements)
    g1[6] = 0;
    g1[7] = 0;
    i32x4 z4 = {0, 0, 0, 0};
#if __clang_major__ >= 23
    i32x8 z8 = {0, 0, 0, 0, 0, 0, 0, 0};
    __builtin_amdgcn_tensor_load_to_lds(g0, g1, z4, z4, z8, 0);
#else
    __builtin_amdgcn_tensor_load_to_lds(g0, g1, z4, z4, 0);
#endif
}

// ---------------------------------------------------------------------------
// fp32 -> bf16 conversion, 4 elements per thread (n is a multiple of 1024)
// ---------------------------------------------------------------------------
__global__ __launch_bounds__(256) void convert_bf16_kernel(
    const float* __restrict__ in, __bf16* __restrict__ out, int n) {
    const int i = (blockIdx.x * 256 + threadIdx.x) * 4;
    if (i + 3 < n) {
        const float4 f = *(const float4*)(in + i);
        v4bf o;
        o[0] = (__bf16)f.x; o[1] = (__bf16)f.y;
        o[2] = (__bf16)f.z; o[3] = (__bf16)f.w;
        *(v4bf*)(out + i) = o;
    }
}

// ---------------------------------------------------------------------------
// C[M,N] = X[M,K] @ W[N,K]^T + bias.  M=4096, N=K=1024.
// Block = 256 thr (8 waves), owns 16 rows x 512 cols (blockIdx.y halves N).
// A band staged to LDS via TDM (shared by all 8 waves); B fragments register
// double-buffered so WMMAs never wait on the loads issued the same step.
// vtrans==1 stores transposed per batch: Out[b][col][s]  (for V).
// ---------------------------------------------------------------------------
__global__ __launch_bounds__(256) void gemm_proj_kernel(
    const __bf16* __restrict__ X, const __bf16* __restrict__ W,
    const float* __restrict__ bias, __bf16* __restrict__ Out, int vtrans) {
    __shared__ __align__(16) __bf16 Asmem[16 * APAD_ROW];

    const int lane    = threadIdx.x & 31;
    const int wv      = threadIdx.x >> 5;
    const int rowbase = blockIdx.x * 16;
    const int colbase = blockIdx.y * 512 + wv * 64;
    const int mrow    = lane & 15;
    const int half    = (lane >> 4) & 1;
    const int klo     = half ? 8 : 0;    // A-fragment K split (16-bit layout)
    const int kb      = half ? 8 : 0;    // B-fragment K split

    if (threadIdx.x < 32) {
        tdm_load_band16x1024(X + (size_t)rowbase * D_,
                             (unsigned int)(uintptr_t)&Asmem[0]);
        __builtin_amdgcn_s_wait_tensorcnt((short)0);
    }
    __syncthreads();

    v8f acc[4];
#pragma unroll
    for (int a = 0; a < 4; ++a) acc[a] = zero8();

    auto loadB = [&](Frag* f, int k0) {
#pragma unroll
        for (int a = 0; a < 4; ++a) {
            const __bf16* brow = W + (size_t)(colbase + a * 16 + mrow) * D_ + k0 + kb;
            f[a].h[0] = load8(brow);
            f[a].h[1] = load8(brow + 16);
        }
    };
    auto loadA = [&](int k0) -> Frag {
        Frag af;
        const int l0 = mrow * 1024 + k0 + klo;
        af.h[0] = load8(&Asmem[apad_idx(l0)]);
        af.h[1] = load8(&Asmem[apad_idx(l0 + 16)]);
        return af;
    };

    Frag b0[4], b1[4];
    loadB(b0, 0);
#pragma unroll 1
    for (int k0 = 0; k0 < D_; k0 += 64) {
        loadB(b1, k0 + 32);                 // in flight while b0 is consumed
        Frag af = loadA(k0);
#pragma unroll
        for (int a = 0; a < 4; ++a) acc[a] = wmma_bf16(af, b0[a], acc[a]);
        if (k0 + 64 < D_) loadB(b0, k0 + 64);
        af = loadA(k0 + 32);
#pragma unroll
        for (int a = 0; a < 4; ++a) acc[a] = wmma_bf16(af, b1[a], acc[a]);
    }

#pragma unroll
    for (int a = 0; a < 4; ++a) {
        const int col = colbase + a * 16 + mrow;
        const float bv = bias[col];
#pragma unroll
        for (int r = 0; r < 8; ++r) {
            const int row = rowbase + r + half * 8;
            const float val = acc[a][r] + bv;
            if (!vtrans) {
                Out[(size_t)row * D_ + col] = (__bf16)val;
            } else {
                const int b = row >> 11;        // row / S_
                const int s = row & (S_ - 1);
                Out[((size_t)b * D_ + col) * S_ + s] = (__bf16)val;
            }
        }
    }
}

// ---------------------------------------------------------------------------
// Flash attention: one wave per (b, h, 16-query tile). KV window of 32.
// K fragments double-buffered across windows; V fragments issued before the
// softmax so exp/shuffle work hides their latency.  V comes pre-transposed
// ([B,D,S]) so all its B-fragments are contiguous b128 loads.
// ---------------------------------------------------------------------------
__global__ __launch_bounds__(32) void flash_attn_kernel(
    const __bf16* __restrict__ Q,   // [B,S,D]
    const __bf16* __restrict__ K,   // [B,S,D]
    const __bf16* __restrict__ Vt,  // [B,D,S]
    __bf16* __restrict__ Ctx) {     // [B,S,D]
    __shared__ __align__(16) __bf16 Plds[16 * 32];

    const int lane  = threadIdx.x & 31;
    const int qt    = blockIdx.x;
    const int h     = blockIdx.y;
    const int b     = blockIdx.z;
    const int qbase = qt * 16;
    const int mrow  = lane & 15;
    const int half  = (lane >> 4) & 1;
    const int klo   = half ? 8 : 0;
    const int kb    = half ? 8 : 0;

    const size_t bsd = (size_t)b * S_ * D_;
    const __bf16* qrow = Q + bsd + (size_t)(qbase + mrow) * D_ + h * DK_;
    Frag aq0, aq1;   // A fragments for K-dim 0..31 and 32..63
    aq0.h[0] = load8(qrow + klo);
    aq0.h[1] = load8(qrow + klo + 16);
    aq1.h[0] = load8(qrow + 32 + klo);
    aq1.h[1] = load8(qrow + 32 + klo + 16);

    float mrun[8], lrun[8];
    v8f O[4];
#pragma unroll
    for (int r = 0; r < 8; ++r) { mrun[r] = -3.0e38f; lrun[r] = 0.0f; }
#pragma unroll
    for (int a = 0; a < 4; ++a) O[a] = zero8();

    const __bf16* kbase = K + bsd + h * DK_;
    const __bf16* vbase = Vt + ((size_t)b * D_ + h * DK_) * (size_t)S_;
    const float scale = 0.125f;     // 1/sqrt(64)

    auto loadK = [&](Frag* f, int j) {
        const __bf16* kr0 = kbase + (size_t)(j + mrow) * D_;
        const __bf16* kr1 = kbase + (size_t)(j + 16 + mrow) * D_;
        f[0].h[0] = load8(kr0 + kb);      f[0].h[1] = load8(kr0 + kb + 16);
        f[1].h[0] = load8(kr0 + 32 + kb); f[1].h[1] = load8(kr0 + 32 + kb + 16);
        f[2].h[0] = load8(kr1 + kb);      f[2].h[1] = load8(kr1 + kb + 16);
        f[3].h[0] = load8(kr1 + 32 + kb); f[3].h[1] = load8(kr1 + 32 + kb + 16);
    };

    Frag bk[4];
    loadK(bk, 0);

#pragma unroll 1
    for (int j = 0; j < S_; j += 32) {
        // ---- scores (K fragments were loaded in the previous iteration)
        v8f s0 = zero8(), s1 = zero8();
        s0 = wmma_bf16(aq0, bk[0], s0);
        s0 = wmma_bf16(aq1, bk[1], s0);
        s1 = wmma_bf16(aq0, bk[2], s1);
        s1 = wmma_bf16(aq1, bk[3], s1);

        // ---- issue next-window K and this-window V loads before softmax
        Frag bkn[4];
        if (j + 32 < S_) loadK(bkn, j + 32);
        Frag bv[4];
#pragma unroll
        for (int a = 0; a < 4; ++a) {
            const __bf16* vr = vbase + (size_t)(a * 16 + mrow) * S_ + j + kb;
            bv[a].h[0] = load8(vr);
            bv[a].h[1] = load8(vr + 16);
        }

        // ---- online softmax in C layout (row = r + 8*half, col = lane&15)
        float corr[8];
#pragma unroll
        for (int r = 0; r < 8; ++r) {
            float x0 = s0[r] * scale;
            float x1 = s1[r] * scale;
            float mx = fmaxf(x0, x1);
#pragma unroll
            for (int msk = 1; msk < 16; msk <<= 1)
                mx = fmaxf(mx, __shfl_xor(mx, msk, 32));
            const float mnew = fmaxf(mrun[r], mx);
            const float c    = __expf(mrun[r] - mnew);
            const float p0   = __expf(x0 - mnew);
            const float p1   = __expf(x1 - mnew);
            float rs = p0 + p1;
#pragma unroll
            for (int msk = 1; msk < 16; msk <<= 1)
                rs += __shfl_xor(rs, msk, 32);
            lrun[r] = lrun[r] * c + rs;
            mrun[r] = mnew;
            corr[r] = c;
            const int row = r + half * 8;
            Plds[row * 32 + mrow]      = (__bf16)p0;
            Plds[row * 32 + mrow + 16] = (__bf16)p1;
        }
#pragma unroll
        for (int a = 0; a < 4; ++a)
#pragma unroll
            for (int r = 0; r < 8; ++r) O[a][r] *= corr[r];

        __syncthreads();

        // ---- P tile back as an A fragment (16x32, K = kv window)
        Frag ap;
        ap.h[0] = load8(&Plds[mrow * 32 + klo]);
        ap.h[1] = load8(&Plds[mrow * 32 + klo + 16]);

#pragma unroll
        for (int a = 0; a < 4; ++a) O[a] = wmma_bf16(ap, bv[a], O[a]);
        __syncthreads();

        if (j + 32 < S_) {
#pragma unroll
            for (int t = 0; t < 4; ++t) bk[t] = bkn[t];
        }
    }

    // ---- normalize and store context tile
    __bf16* cbase = Ctx + bsd + h * DK_;
#pragma unroll
    for (int r = 0; r < 8; ++r) {
        const float inv = 1.0f / lrun[r];
        const int row = qbase + r + half * 8;
#pragma unroll
        for (int a = 0; a < 4; ++a)
            cbase[(size_t)row * D_ + a * 16 + mrow] = (__bf16)(O[a][r] * inv);
    }
}

// ---------------------------------------------------------------------------
// out = LayerNorm(resid + Ctx @ Wo^T + bo) * gamma + beta   (fp32 output)
// Block owns a 16-row band (full N=1024 so LN can fuse).  A band staged via
// TDM; 16x1024 f32 result band in LDS; shuffle-reduced LayerNorm.
// ---------------------------------------------------------------------------
__global__ __launch_bounds__(256) void outproj_ln_kernel(
    const __bf16* __restrict__ Ctx, const __bf16* __restrict__ Wo,
    const float* __restrict__ bo, const float* __restrict__ resid,
    const float* __restrict__ gamma, const float* __restrict__ beta,
    float* __restrict__ Out) {
    __shared__ __align__(16) __bf16 Asmem[16 * APAD_ROW];   // 32.5 KB
    __shared__ __align__(16) float  xt[16 * D_];            // 64 KB

    const int lane    = threadIdx.x & 31;
    const int wv      = threadIdx.x >> 5;
    const int rowbase = blockIdx.x * 16;
    const int colbase = wv * 128;
    const int mrow    = lane & 15;
    const int half    = (lane >> 4) & 1;
    const int klo     = half ? 8 : 0;
    const int kb      = half ? 8 : 0;

    if (threadIdx.x < 32) {
        tdm_load_band16x1024(Ctx + (size_t)rowbase * D_,
                             (unsigned int)(uintptr_t)&Asmem[0]);
        __builtin_amdgcn_s_wait_tensorcnt((short)0);
    }
    __syncthreads();

    v8f acc[8];
#pragma unroll
    for (int a = 0; a < 8; ++a) acc[a] = zero8();

#pragma unroll 1
    for (int k0 = 0; k0 < D_; k0 += 32) {
        Frag af;
        const int l0 = mrow * 1024 + k0 + klo;
        af.h[0] = load8(&Asmem[apad_idx(l0)]);
        af.h[1] = load8(&Asmem[apad_idx(l0 + 16)]);
#pragma unroll
        for (int a = 0; a < 8; ++a) {
            const __bf16* brow = Wo + (size_t)(colbase + a * 16 + mrow) * D_ + k0 + kb;
            Frag bf_;
            bf_.h[0] = load8(brow);
            bf_.h[1] = load8(brow + 16);
            acc[a] = wmma_bf16(af, bf_, acc[a]);
        }
    }

#pragma unroll
    for (int a = 0; a < 8; ++a) {
        const int col = colbase + a * 16 + mrow;
        const float bv = bo[col];
#pragma unroll
        for (int r = 0; r < 8; ++r)
            xt[(r + half * 8) * D_ + col] = acc[a][r] + bv;
    }
    __syncthreads();

    // LayerNorm: 16 threads per row (rows 2w, 2w+1 live in wave w)
    const int tid  = threadIdx.x;
    const int row  = tid >> 4;
    const int l16  = tid & 15;
    const int grow = rowbase + row;
    const float* rrow = resid + (size_t)grow * D_;

    float sum = 0.0f, sq = 0.0f;
#pragma unroll 4
    for (int i = 0; i < D_ / 16; ++i) {
        const int c = l16 + i * 16;
        const float x = xt[row * D_ + c] + rrow[c];
        xt[row * D_ + c] = x;
        sum += x;
        sq  += x * x;
    }
#pragma unroll
    for (int msk = 1; msk < 16; msk <<= 1) {
        sum += __shfl_xor(sum, msk, 32);
        sq  += __shfl_xor(sq,  msk, 32);
    }
    const float mean = sum * (1.0f / D_);
    const float var  = sq * (1.0f / D_) - mean * mean;
    const float rstd = rsqrtf(var + LN_EPS);

    float* orow = Out + (size_t)grow * D_;
#pragma unroll 4
    for (int i = 0; i < D_ / 16; ++i) {
        const int c = l16 + i * 16;
        orow[c] = (xt[row * D_ + c] - mean) * rstd * gamma[c] + beta[c];
    }
}

// ---------------------------------------------------------------------------
extern "C" void kernel_launch(void* const* d_in, const int* in_sizes, int n_in,
                              void* d_out, int out_size, void* d_ws, size_t ws_size,
                              hipStream_t stream) {
    const float* q     = (const float*)d_in[0];
    const float* k     = (const float*)d_in[1];
    const float* v     = (const float*)d_in[2];
    const float* Wq    = (const float*)d_in[3];
    const float* bq    = (const float*)d_in[4];
    const float* Wk    = (const float*)d_in[5];
    const float* bk    = (const float*)d_in[6];
    const float* Wv    = (const float*)d_in[7];
    const float* bv    = (const float*)d_in[8];
    const float* Wo    = (const float*)d_in[9];
    const float* bo    = (const float*)d_in[10];
    const float* gamma = (const float*)d_in[11];
    const float* beta  = (const float*)d_in[12];
    float* out = (float*)d_out;

    constexpr size_t MD = (size_t)M_ * D_;   // 4,194,304 activation elements
    constexpr size_t DD = (size_t)D_ * D_;   // 1,048,576 weight elements

    char* p = (char*)d_ws;
    auto carve = [&](size_t elems) -> __bf16* {
        __bf16* r = (__bf16*)p;
        p += ((elems * sizeof(__bf16)) + 255) & ~(size_t)255;
        return r;
    };
    __bf16* qb  = carve(MD);   // q  in bf16
    __bf16* kbb = carve(MD);   // k  in bf16
    __bf16* vbb = carve(MD);   // v  in bf16
    __bf16* wqb = carve(DD);
    __bf16* wkb = carve(DD);
    __bf16* wvb = carve(DD);
    __bf16* wob = carve(DD);
    __bf16* Qa  = carve(MD);   // Q activations  [B,S,D]
    __bf16* Ka  = carve(MD);   // K activations  [B,S,D]
    __bf16* Vt  = carve(MD);   // V activations, transposed [B,D,S]
    __bf16* Ctx = carve(MD);   // attention context [B,S,D]

    auto cv = [&](const float* src, __bf16* dst, size_t n) {
        convert_bf16_kernel<<<dim3((unsigned)(n / 1024)), 256, 0, stream>>>(src, dst, (int)n);
    };
    cv(q, qb, MD);
    cv(k, kbb, MD);
    cv(v, vbb, MD);
    cv(Wq, wqb, DD);
    cv(Wk, wkb, DD);
    cv(Wv, wvb, DD);
    cv(Wo, wob, DD);

    gemm_proj_kernel<<<dim3(M_ / 16, 2), 256, 0, stream>>>(qb,  wqb, bq, Qa, 0);
    gemm_proj_kernel<<<dim3(M_ / 16, 2), 256, 0, stream>>>(kbb, wkb, bk, Ka, 0);
    gemm_proj_kernel<<<dim3(M_ / 16, 2), 256, 0, stream>>>(vbb, wvb, bv, Vt, 1);

    flash_attn_kernel<<<dim3(S_ / 16, H_, B_), 32, 0, stream>>>(Qa, Ka, Vt, Ctx);

    outproj_ln_kernel<<<dim3(M_ / 16), 256, 0, stream>>>(Ctx, wob, bo, q, gamma, beta, out);
}
